// AE_spikes_18433999635174
// MI455X (gfx1250) — matrix operand order repaired and checked by reading, not compiled, not run on hardware
//
#include <hip/hip_runtime.h>

typedef _Float16 h16;
typedef __attribute__((ext_vector_type(16))) _Float16     v16h;
typedef __attribute__((ext_vector_type(8)))  float        v8f;
typedef __attribute__((ext_vector_type(4)))  unsigned int v4u;

union FragH {
  v16h h;
  v4u  q[2];
  unsigned short s[16];
};

#define BATCH   16384
#define IN_DIM  784
#define KPAD    800      // 784 padded to a multiple of 32 (25 K-chunks)
#define HID     128
#define TSTEPS  16
#define MTILE   64       // batch rows per block (4 waves x 16 rows)

// Raw 32-bit LDS byte offset of a generic pointer into dynamic shared memory.
__device__ __forceinline__ unsigned lds_offset(const void* p) {
  return (unsigned)(unsigned long long)(__attribute__((address_space(3))) const void*)p;
}

// ---------------------------------------------------------------------------
// Precompute per-feature 16-bit spike masks (input "membrane" is a Bresenham
// pattern since act = k/16 is constant across steps).
// ---------------------------------------------------------------------------
__global__ void prep_mask_kernel(const float* __restrict__ feats,
                                 const float* __restrict__ in_bins0,
                                 unsigned short* __restrict__ mask) {
  int i = blockIdx.x * blockDim.x + threadIdx.x;
  if (i >= BATCH * KPAD) return;
  int row = i / KPAD;
  int f   = i - row * KPAD;
  unsigned m = 0;
  if (f < IN_DIM) {
    float lo   = in_bins0[0];
    float invh = 1.0f / (in_bins0[1] - in_bins0[0]);
    float x = feats[row * IN_DIM + f];
    int k = (int)floorf((x - lo) * invh);       // digitize(...) - 1
    k = k < 0 ? 0 : (k > TSTEPS ? TSTEPS : k);
#pragma unroll
    for (int t = 0; t < TSTEPS; ++t) {
      unsigned bit = (unsigned)(((t + 1) * k) >> 4) - (unsigned)((t * k) >> 4);
      m |= bit << t;
    }
  }
  mask[i] = (unsigned short)m;
}

// ---------------------------------------------------------------------------
// Pre-swizzle W[N][Ktrue] (row-major) * (h_in/h_out) into WMMA B-operand
// fragment images: fragment (kc, jt) = 32 lanes x 16 halves contiguous.
// CDNA5 f16 B layout (32x16): col n = jt*16 + lane%16; half h -> K =
// kc*32 + (lane/16)*8 + h (h<8), kc*32 + 16 + (lane/16)*8 + h-8 (h>=8).  K >= Ktrue is zero-padded.
// ---------------------------------------------------------------------------
__global__ void swz_kernel(const float* __restrict__ W,
                           const float* __restrict__ in_bins,
                           const float* __restrict__ out_bins,
                           h16* __restrict__ dst,
                           int N, int Ktrue, int JT, int total) {
  int i = blockIdx.x * blockDim.x + threadIdx.x;
  if (i >= total) return;
  float scale = (in_bins[1] - in_bins[0]) / (out_bins[1] - out_bins[0]);
  int h    = i & 15;
  int lane = (i >> 4) & 31;
  int frag = i >> 9;
  int jt   = frag % JT;
  int kc   = frag / JT;
  int n = jt * 16 + (lane & 15);
  int k = kc * 32 + ((lane >> 4) << 3) + (h & 7) + ((h >> 3) << 4);
  float v = (k < Ktrue && n < N) ? scale * W[n * Ktrue + k] : 0.0f;
  dst[i] = (h16)v;
}

// ---------------------------------------------------------------------------
// One hidden layer (K = N = 128): acc = membrane; acc += spk_in @ Weffᵀ;
// spike; write back membrane and fp16 spikes.  Weights read from LDS
// (async-staged); everything is wave-private (each wave owns 16 rows).
// ---------------------------------------------------------------------------
__device__ __forceinline__ void hid_layer(const h16* WLds,
                                          float* tmem,
                                          const h16* spkIn, h16* spkOut,
                                          int wr0, int g, int mn, int lane) {
  const int arow = wr0 + mn;
  v8f acc[8];
#pragma unroll
  for (int j = 0; j < 8; ++j)
#pragma unroll
    for (int r = 0; r < 8; ++r)
      acc[j][r] = tmem[(wr0 + r + 8 * g) * HID + j * 16 + mn];

  FragH a[4];
#pragma unroll
  for (int c = 0; c < 4; ++c) {   // A f16 layout: halves 0..7->K=g*8+h, 8..15->K=16+g*8+h
    a[c].q[0] = *(const v4u*)(spkIn + arow * HID + c * 32 + g * 8);
    a[c].q[1] = *(const v4u*)(spkIn + arow * HID + c * 32 + 16 + g * 8);
  }
#pragma unroll
  for (int j = 0; j < 8; ++j) {
#pragma unroll
    for (int c = 0; c < 4; ++c) {
      FragH b;
      const h16* bp = WLds + (((c * 8 + j) * 32 + lane) << 4);
      b.q[0] = *(const v4u*)(bp);
      b.q[1] = *(const v4u*)(bp + 8);
      acc[j] = __builtin_amdgcn_wmma_f32_16x16x32_f16(false, a[c].h, false, b.h,
                                                      (short)0, acc[j], false, false);
    }
  }
#pragma unroll
  for (int j = 0; j < 8; ++j)
#pragma unroll
    for (int r = 0; r < 8; ++r) {
      float v = acc[j][r];
      float s = (v >= 1.0f) ? 1.0f : 0.0f;
      int idx = (wr0 + r + 8 * g) * HID + j * 16 + mn;
      tmem[idx]   = v - s;
      spkOut[idx] = (h16)s;
    }
}

// ---------------------------------------------------------------------------
// Main SNN forward: 16 time steps, 4 layers, all matmuls via v_wmma.
// Block = 128 threads (4 waves); each wave owns 16 batch rows end-to-end.
// ---------------------------------------------------------------------------
__global__ __launch_bounds__(128, 2) void snn_fwd_kernel(
    const unsigned short* __restrict__ mask,
    const h16* __restrict__ W0s, const h16* __restrict__ W1s,
    const h16* __restrict__ W2s, const h16* __restrict__ W3s,
    const float* __restrict__ b0, const float* __restrict__ b1,
    const float* __restrict__ b2, const float* __restrict__ b3,
    const float* __restrict__ ib0, const float* __restrict__ ob0,
    const float* __restrict__ ib1, const float* __restrict__ ob1,
    const float* __restrict__ ib2, const float* __restrict__ ob2,
    const float* __restrict__ ib3, const float* __restrict__ ob3,
    float* __restrict__ t3ws, float* __restrict__ out) {
  extern __shared__ char smem[];
  float* t0 = (float*)smem;                 // [MTILE][128] f32
  float* t1 = t0 + MTILE * HID;
  float* t2 = t1 + MTILE * HID;
  h16* spkA = (h16*)(t2 + MTILE * HID);     // [MTILE][128] f16
  h16* spkB = spkA + MTILE * HID;
  h16* W1L  = spkB + MTILE * HID;           // 32KB swizzled W1 fragments
  h16* W2L  = W1L + 4 * 8 * 512;            // 32KB swizzled W2 fragments

  const int tid  = threadIdx.x;
  const int wave = tid >> 5;
  const int lane = tid & 31;
  const int g    = lane >> 4;
  const int mn   = lane & 15;
  const int wr0  = wave << 4;               // wave's local row base
  const int arow = wr0 + mn;
  const int row0 = blockIdx.x * MTILE;

  // Async-stage W1/W2 fragment images into LDS (shared by all 4 waves).
  {
    unsigned l1 = lds_offset(W1L);
    unsigned l2 = lds_offset(W2L);
#pragma unroll 1
    for (int it = 0; it < 16; ++it) {
      unsigned off = (unsigned)(tid + it * 128) * 16u;   // bytes, 16B each
      asm volatile("global_load_async_to_lds_b128 %0, %1, %2 offset:0"
                   :: "v"(l1 + off), "v"(off), "s"(W1s) : "memory");
      asm volatile("global_load_async_to_lds_b128 %0, %1, %2 offset:0"
                   :: "v"(l2 + off), "v"(off), "s"(W2s) : "memory");
    }
  }

  const float s0 = (ib0[1] - ib0[0]) / (ob0[1] - ob0[0]);
  const float s1 = (ib1[1] - ib1[0]) / (ob1[1] - ob1[0]);
  const float s2 = (ib2[1] - ib2[0]) / (ob2[1] - ob2[0]);
  const float s3 = (ib3[1] - ib3[0]) / (ob3[1] - ob3[0]);
  const float h3 = ob3[1] - ob3[0];

  // Membrane init = effective bias (reset_totals).  Wave-private rows.
  for (int e = lane; e < 16 * HID; e += 32) {
    int n = e & (HID - 1);
    int idx = (wr0 + (e >> 7)) * HID + n;
    t0[idx] = s0 * b0[n];
    t1[idx] = s1 * b1[n];
    t2[idx] = s2 * b2[n];
  }
  for (int e = lane; e < 16 * IN_DIM; e += 32) {
    int r = e / IN_DIM;
    int n = e - r * IN_DIM;
    t3ws[(size_t)(row0 + wr0 + r) * IN_DIM + n] = s3 * b3[n];
  }
  asm volatile("s_wait_asynccnt 0x0" ::: "memory");
  __threadfence();
  __syncthreads();

  const unsigned short* mrow = mask + (size_t)(row0 + arow) * KPAD;

#pragma unroll 1
  for (int t = 0; t < TSTEPS; ++t) {
    // ---------------- layer 0: z = s_in @ W0effᵀ  (K = 800 padded) ----------
    v8f acc[8];
#pragma unroll
    for (int j = 0; j < 8; ++j)
#pragma unroll
      for (int r = 0; r < 8; ++r)
        acc[j][r] = t0[(wr0 + r + 8 * g) * HID + j * 16 + mn];

#pragma unroll 1
    for (int c = 0; c < KPAD / 32; ++c) {
      FragH a;  // build s_in A-fragment from spike masks, fully in registers
      v4u m0v = *(const v4u*)(mrow + c * 32 + g * 8);
      v4u m1v = *(const v4u*)(mrow + c * 32 + 16 + g * 8);
#pragma unroll
      for (int h = 0; h < 8; ++h) {
        unsigned w0 = (h & 1) ? (m0v[h >> 1] >> 16) : (m0v[h >> 1] & 0xffffu);
        unsigned w1 = (h & 1) ? (m1v[h >> 1] >> 16) : (m1v[h >> 1] & 0xffffu);
        a.s[h]     = ((w0 >> t) & 1u) ? (unsigned short)0x3C00 : (unsigned short)0;
        a.s[h + 8] = ((w1 >> t) & 1u) ? (unsigned short)0x3C00 : (unsigned short)0;
      }
#pragma unroll
      for (int j = 0; j < 8; ++j) {
        FragH b;
        const h16* bp = W0s + (((c * 8 + j) * 32 + lane) << 4);
        b.q[0] = *(const v4u*)(bp);
        b.q[1] = *(const v4u*)(bp + 8);
        acc[j] = __builtin_amdgcn_wmma_f32_16x16x32_f16(false, a.h, false, b.h,
                                                        (short)0, acc[j], false, false);
      }
    }
#pragma unroll
    for (int j = 0; j < 8; ++j)
#pragma unroll
      for (int r = 0; r < 8; ++r) {
        float v = acc[j][r];
        float s = (v >= 1.0f) ? 1.0f : 0.0f;
        int idx = (wr0 + r + 8 * g) * HID + j * 16 + mn;
        t0[idx]   = v - s;
        spkA[idx] = (h16)s;
      }
    __syncthreads();

    // ---------------- layers 1,2 (weights resident in LDS) ----------------
    hid_layer(W1L, t1, spkA, spkB, wr0, g, mn, lane);
    __syncthreads();
    hid_layer(W2L, t2, spkB, spkA, wr0, g, mn, lane);
    __syncthreads();

    // ---------------- layer 3: N = 784, membrane in global ws --------------
    FragH a3[4];
#pragma unroll
    for (int c = 0; c < 4; ++c) {
      a3[c].q[0] = *(const v4u*)(spkA + arow * HID + c * 32 + g * 8);
      a3[c].q[1] = *(const v4u*)(spkA + arow * HID + c * 32 + 16 + g * 8);
    }
#pragma unroll 1
    for (int j = 0; j < IN_DIM / 16; ++j) {
      v8f acc3;
#pragma unroll
      for (int r = 0; r < 8; ++r)
        acc3[r] = t3ws[(size_t)(row0 + wr0 + r + 8 * g) * IN_DIM + j * 16 + mn];
#pragma unroll
      for (int c = 0; c < 4; ++c) {
        FragH b;
        const h16* bp = W3s + (((c * (IN_DIM / 16) + j) * 32 + lane) << 4);
        b.q[0] = *(const v4u*)(bp);
        b.q[1] = *(const v4u*)(bp + 8);
        acc3 = __builtin_amdgcn_wmma_f32_16x16x32_f16(false, a3[c].h, false, b.h,
                                                      (short)0, acc3, false, false);
      }
#pragma unroll
      for (int r = 0; r < 8; ++r) {
        float v = acc3[r];
        float s = (v >= 1.0f) ? 1.0f : 0.0f;
        size_t gidx = (size_t)(row0 + wr0 + r + 8 * g) * IN_DIM + j * 16 + mn;
        t3ws[gidx] = v - s;
        if (t == 0)              out[gidx] = s;               // overwrite (poison-safe)
        else if (t == TSTEPS - 1) out[gidx] = (out[gidx] + s) * h3;  // reconstruct
        else                      out[gidx] += s;
      }
    }
    __syncthreads();
  }
}

// ---------------------------------------------------------------------------
// Workspace layout (bytes, 256-aligned):
//   mask16 : 16384*800*2        = 26,214,400
//   W0s    : 25*8*512 halves    =    204,800
//   W1s    : 4*8*512            =     32,768
//   W2s    :                    =     32,768
//   W3s    : 4*49*512           =    200,704
//   t3ws   : 16384*784*4        = 51,380,224
//   total ≈ 78.1 MB
// ---------------------------------------------------------------------------
extern "C" void kernel_launch(void* const* d_in, const int* in_sizes, int n_in,
                              void* d_out, int out_size, void* d_ws, size_t ws_size,
                              hipStream_t stream) {
  (void)in_sizes; (void)n_in; (void)out_size; (void)ws_size;
  const float* feats = (const float*)d_in[0];
  const float* W0 = (const float*)d_in[1];
  const float* b0 = (const float*)d_in[2];
  const float* W1 = (const float*)d_in[3];
  const float* b1 = (const float*)d_in[4];
  const float* W2 = (const float*)d_in[5];
  const float* b2 = (const float*)d_in[6];
  const float* W3 = (const float*)d_in[7];
  const float* b3 = (const float*)d_in[8];
  const float* ib0 = (const float*)d_in[9];
  const float* ob0 = (const float*)d_in[10];
  const float* ib1 = (const float*)d_in[11];
  const float* ob1 = (const float*)d_in[12];
  const float* ib2 = (const float*)d_in[13];
  const float* ob2 = (const float*)d_in[14];
  const float* ib3 = (const float*)d_in[15];
  const float* ob3 = (const float*)d_in[16];

  char* ws = (char*)d_ws;
  unsigned short* mask = (unsigned short*)(ws);
  h16* W0s  = (h16*)(ws + 26214400);
  h16* W1s  = (h16*)(ws + 26419200);
  h16* W2s  = (h16*)(ws + 26451968);
  h16* W3s  = (h16*)(ws + 26484736);
  float* t3 = (float*)(ws + 26685440);

  hipLaunchKernelGGL(prep_mask_kernel, dim3((BATCH * KPAD + 255) / 256), dim3(256),
                     0, stream, feats, ib0, mask);
  hipLaunchKernelGGL(swz_kernel, dim3((102400 + 255) / 256), dim3(256), 0, stream,
                     W0, ib0, ob0, W0s, HID, IN_DIM, 8, 102400);
  hipLaunchKernelGGL(swz_kernel, dim3((16384 + 255) / 256), dim3(256), 0, stream,
                     W1, ib1, ob1, W1s, HID, HID, 8, 16384);
  hipLaunchKernelGGL(swz_kernel, dim3((16384 + 255) / 256), dim3(256), 0, stream,
                     W2, ib2, ob2, W2s, HID, HID, 8, 16384);
  hipLaunchKernelGGL(swz_kernel, dim3((100352 + 255) / 256), dim3(256), 0, stream,
                     W3, ib3, ob3, W3s, IN_DIM, HID, 49, 100352);

  // LDS: 3 membranes (f32) + 2 spike buffers (f16) + 2 weight images (f16)
  const size_t lds_bytes = (size_t)(3 * MTILE * HID * 4 + 2 * MTILE * HID * 2
                                    + 2 * 4 * 8 * 512 * 2);   // 196608
  hipLaunchKernelGGL(snn_fwd_kernel, dim3(BATCH / MTILE), dim3(128), lds_bytes, stream,
                     mask, W0s, W1s, W2s, W3s, b0, b1, b2, b3,
                     ib0, ob0, ib1, ob1, ib2, ob2, ib3, ob3,
                     t3, (float*)d_out);
}
